// LMUNet_88725434400860
// MI455X (gfx1250) — matrix operand order repaired and checked by reading, not compiled, not run on hardware
//
#include <hip/hip_runtime.h>
#include <hip/hip_bf16.h>

// ---------------- problem constants (match reference) ----------------
#define T_STEPS 40
#define B_SZ    1024
#define I_SZ    128
#define H_SZ    1024
#define M_SZ    256
#define O_SZ    128
#define K_TOT   (I_SZ + H_SZ + M_SZ)   // 1408
#define NCHUNK  (K_TOT / 32)           // 44

typedef __bf16 bf16;
typedef __attribute__((ext_vector_type(16))) __bf16 bf16x16;
typedef __attribute__((ext_vector_type(8)))  __bf16 bf16x8;
typedef __attribute__((ext_vector_type(8)))  float  f32x8;
typedef __attribute__((ext_vector_type(2)))  float  f32x2;
typedef __attribute__((ext_vector_type(4)))  unsigned int u32x4;
typedef __attribute__((ext_vector_type(8)))  int i32x8;
typedef __attribute__((ext_vector_type(4)))  int i32x4;

// ---------------- WMMA helpers (wave32) ----------------
__device__ __forceinline__ f32x8 wmma_bf16(bf16x16 a, bf16x16 b, f32x8 c) {
    return __builtin_amdgcn_wmma_f32_16x16x32_bf16(false, a, false, b, (short)0, c, false, false);
}

// Build a 16-element fragment from two 16-byte halves.
__device__ __forceinline__ bf16x16 frag_from(const bf16* p0, const bf16* p1) {
    bf16x8 lo = *(const bf16x8*)p0;
    bf16x8 hi = *(const bf16x8*)p1;
    return __builtin_shufflevector(lo, hi, 0,1,2,3,4,5,6,7,8,9,10,11,12,13,14,15);
}

// ---------------- TDM: DMA one 128-row x 32-col bf16 tile into LDS ----------------
// LDS layout: 128 rows of 32 data elements + 8 pad elements (row stride 80B,
// conflict-free for the WMMA fragment reads). TDM pads in hardware:
// pad_interval = 16 DWORDs (one 64B row), pad_amount = 4 DWORDs (16B).
#define RFL(v) __builtin_amdgcn_readfirstlane((int)(v))
__device__ __forceinline__ void tdm_load_tile(const bf16* gsrc, int ld_elems,
                                              unsigned lds_byte_off) {
    const unsigned long long ga = (unsigned long long)(uintptr_t)gsrc;
    u32x4 g0;
    g0[0] = (unsigned)RFL(1u);                                  // count=1 (user D#)
    g0[1] = (unsigned)RFL(lds_byte_off);                        // lds_addr
    g0[2] = (unsigned)RFL((unsigned)(ga & 0xffffffffu));        // global_addr[31:0]
    g0[3] = (unsigned)RFL(((unsigned)((ga >> 32) & 0x1ffffffu)) // global_addr[56:32]
                          | (2u << 30));                        // type = 2 (image)
    const unsigned td0 = (unsigned)ld_elems;                    // tensor_dim0
    const unsigned td1 = 128u;                                  // tensor_dim1
    i32x8 g1;
    g1[0] = RFL((1u << 16) |      // data_size = 2 bytes
                (1u << 20) |      // pad_enable
                (3u << 22) |      // pad_interval: 16 DWORDs
                (3u << 25));      // pad_amount : 4 DWORDs
    g1[1] = RFL((td0 & 0xffffu) << 16);                         // tensor_dim0[15:0]
    g1[2] = RFL(((td0 >> 16) & 0xffffu) | ((td1 & 0xffffu) << 16));
    g1[3] = RFL(((td1 >> 16) & 0xffffu) | (32u << 16));         // tile_dim0 = 32
    g1[4] = RFL(128u);                                          // tile_dim1=128, tile_dim2=0
    g1[5] = RFL(td0);                                           // tensor_dim0_stride[31:0]
    g1[6] = RFL(0);                                             // stride hi / dim1_stride
    g1[7] = RFL(0);
    const i32x4 z4 = {0, 0, 0, 0};
    const i32x8 z8 = {0, 0, 0, 0, 0, 0, 0, 0};
    // 6-arg form (this toolchain): (g0, g1, g2, g3, g4, cpol)
    __builtin_amdgcn_tensor_load_to_lds(g0, g1, z4, z4, z8, 0);
}

// ---------------- prep kernels ----------------
__global__ __launch_bounds__(256) void k_f32_to_bf16(const float* __restrict__ s,
                                                     bf16* __restrict__ d, int n) {
    int i = blockIdx.x * 256 + threadIdx.x;
    if (i < n) d[i] = (bf16)s[i];
}
__global__ __launch_bounds__(256) void k_zero_bf16(bf16* __restrict__ d, int n) {
    int i = blockIdx.x * 256 + threadIdx.x;
    if (i < n) d[i] = (bf16)0.0f;
}
__global__ __launch_bounds__(256) void k_zero_f32(float* __restrict__ d, int n) {
    int i = blockIdx.x * 256 + threadIdx.x;
    if (i < n) d[i] = 0.0f;
}

// ---------------- fused u + memory update (f32 WMMA 16x16x4) ----------------
__global__ __launch_bounds__(512) void k_um(int t,
        const float* __restrict__ x, const bf16* __restrict__ h_prev,
        const float* __restrict__ m_prev,
        const float* __restrict__ ex, const float* __restrict__ eh,
        const float* __restrict__ em,
        const float* __restrict__ Ad, const float* __restrict__ Bd,
        float* __restrict__ m_next, bf16* __restrict__ m_bf) {
    __shared__ float u_s[16];
    const int lane = threadIdx.x & 31;
    const int wave = threadIdx.x >> 5;        // 0..15
    const int b0 = blockIdx.x * 16;

    { // phase 1: one wave per batch row
        const int b = b0 + wave;
        const float* xr = x + (size_t)t * B_SZ * I_SZ + (size_t)b * I_SZ;
        const bf16*  hr = h_prev + (size_t)b * H_SZ;
        const float* mr = m_prev + (size_t)b * M_SZ;
        float s = 0.0f;
        #pragma unroll
        for (int j = lane; j < I_SZ; j += 32) s += xr[j] * ex[j];
        for (int j = lane; j < H_SZ; j += 32) s += (float)hr[j] * eh[j];
        #pragma unroll
        for (int j = lane; j < M_SZ; j += 32) s += mr[j] * em[j];
        #pragma unroll
        for (int off = 16; off > 0; off >>= 1) s += __shfl_xor(s, off, 32);
        if (lane == 0) u_s[wave] = s;
    }
    __syncthreads();

    // phase 2: rows b0..b0+15, cols n0 = wave*16 (M = 256 = 16 tiles)
    const int laneLo = lane & 15, laneHi = lane >> 4;
    const int n0 = wave * 16;
    f32x8 acc = {};
    for (int kc = 0; kc < M_SZ; kc += 4) {
        f32x2 a = *(const f32x2*)(m_prev + (size_t)(b0 + laneLo) * M_SZ + kc + laneHi * 2);
        f32x2 b = *(const f32x2*)(Ad + (size_t)(n0 + laneLo) * M_SZ + kc + laneHi * 2);
        acc = __builtin_amdgcn_wmma_f32_16x16x4_f32(false, a, false, b, (short)0, acc, false, false);
    }
    #pragma unroll
    for (int r = 0; r < 8; r++) {
        const int row = b0 + laneHi * 8 + r;
        const int col = n0 + laneLo;
        const float v = acc[r] + u_s[laneHi * 8 + r] * Bd[col];
        m_next[(size_t)row * M_SZ + col] = v;
        m_bf  [(size_t)row * M_SZ + col] = (bf16)v;
    }
}

// ---------------- h_cur = tanh([x_t | h_prev | m] @ [Wx|Wh|Wm]^T) ----------------
// 128x128 block tile, 8 waves. Per 32-wide k-chunk the A (activations) and B
// (weights) tiles are DMA'd into double-buffered LDS by the Tensor Data Mover
// (issued by wave 0, tracked with TENSORcnt) while the previous chunk's WMMAs run.
__global__ __launch_bounds__(256) void k_hidden(int t,
        const bf16* __restrict__ x_bf, const bf16* __restrict__ h_prev,
        const bf16* __restrict__ m_bf,
        const bf16* __restrict__ Wx, const bf16* __restrict__ Wh,
        const bf16* __restrict__ Wm, bf16* __restrict__ h_cur) {
    __shared__ __align__(16) bf16 aT[2][128][40];  // 32 data + 8 pad per row
    __shared__ __align__(16) bf16 bT[2][128][40];
    const int tid  = threadIdx.x;
    const int lane = tid & 31;
    const int wave = tid >> 5;
    const int laneLo = lane & 15, laneHi = lane >> 4;
    const int rBlk = blockIdx.y * 128;             // batch-row block
    const int nBlk = blockIdx.x * 128;             // hidden-col block
    const int rW = (wave >> 2) * 64;               // wave row offset in block
    const int nW = (wave & 3) * 32;                // wave col offset in block
    const bf16* xcur = x_bf + (size_t)t * B_SZ * I_SZ;

    auto seg = [&](int kc, const bf16*& aS, const bf16*& bS, int& ld, int& kl) {
        if (kc < I_SZ)             { aS = xcur;   bS = Wx; ld = I_SZ; kl = kc; }
        else if (kc < I_SZ + H_SZ) { aS = h_prev; bS = Wh; ld = H_SZ; kl = kc - I_SZ; }
        else                       { aS = m_bf;   bS = Wm; ld = M_SZ; kl = kc - I_SZ - H_SZ; }
    };

    // prologue: DMA chunk 0 into buffer 0
    if (wave == 0) {
        const bf16 *aS, *bS; int ld, kl;
        seg(0, aS, bS, ld, kl);
        tdm_load_tile(aS + (size_t)rBlk * ld + kl, ld,
                      (unsigned)(uintptr_t)&aT[0][0][0]);
        tdm_load_tile(bS + (size_t)nBlk * ld + kl, ld,
                      (unsigned)(uintptr_t)&bT[0][0][0]);
    }

    f32x8 acc[4][2] = {};
    for (int ci = 0; ci < NCHUNK; ci++) {
        const int cur = ci & 1, nxt = cur ^ 1;
        if (wave == 0) __builtin_amdgcn_s_wait_tensorcnt(0);
        __syncthreads();                       // tile ready; prev reads of nxt done
        if (wave == 0 && ci + 1 < NCHUNK) {    // overlap next DMA with compute
            const bf16 *aS, *bS; int ld, kl;
            seg((ci + 1) * 32, aS, bS, ld, kl);
            tdm_load_tile(aS + (size_t)rBlk * ld + kl, ld,
                          (unsigned)(uintptr_t)&aT[nxt][0][0]);
            tdm_load_tile(bS + (size_t)nBlk * ld + kl, ld,
                          (unsigned)(uintptr_t)&bT[nxt][0][0]);
        }
        bf16x16 bfr[2];
        #pragma unroll
        for (int j = 0; j < 2; j++) {
            const bf16* p = &bT[cur][nW + 16 * j + laneLo][laneHi * 16];
            bfr[j] = frag_from(p, p + 8);      // K contiguous per lane
        }
        #pragma unroll
        for (int i = 0; i < 4; i++) {
            const bf16* p = &aT[cur][rW + 16 * i + laneLo][laneHi * 16];
            bf16x16 afr = frag_from(p, p + 8); // K contiguous per lane
            #pragma unroll
            for (int j = 0; j < 2; j++) acc[i][j] = wmma_bf16(afr, bfr[j], acc[i][j]);
        }
    }

    #pragma unroll
    for (int i = 0; i < 4; i++)
        #pragma unroll
        for (int j = 0; j < 2; j++)
            #pragma unroll
            for (int r = 0; r < 8; r++) {
                const int row = rBlk + rW + 16 * i + laneHi * 8 + r;
                const int col = nBlk + nW + 16 * j + laneLo;
                h_cur[(size_t)row * H_SZ + col] = (bf16)tanhf(acc[i][j][r]);
            }
}

// ---------------- out = hbase(rows x H) @ Wo^T, rows given by grid ----------------
__global__ __launch_bounds__(256) void k_out(
        const bf16* __restrict__ hbase, const bf16* __restrict__ Wo,
        float* __restrict__ out) {
    const int lane = threadIdx.x & 31;
    const int wave = threadIdx.x >> 5;
    const int laneLo = lane & 15, laneHi = lane >> 4;
    const int rB = blockIdx.x * 128 + (wave >> 2) * 64;
    const int nB = (wave & 3) * 32;                // O = 128 across 4 wave cols
    f32x8 acc[4][2] = {};
    for (int kc = 0; kc < H_SZ; kc += 32) {
        bf16x16 bfr[2];
        #pragma unroll
        for (int j = 0; j < 2; j++) {
            const bf16* p = Wo + (size_t)(nB + 16 * j + laneLo) * H_SZ + kc + laneHi * 16;
            bfr[j] = frag_from(p, p + 8);
        }
        #pragma unroll
        for (int i = 0; i < 4; i++) {
            const bf16* p = hbase + (size_t)(rB + 16 * i + laneLo) * H_SZ + kc + laneHi * 16;
            __builtin_prefetch(p + 32, 0, 0);
            bf16x16 afr = frag_from(p, p + 8);
            #pragma unroll
            for (int j = 0; j < 2; j++) acc[i][j] = wmma_bf16(afr, bfr[j], acc[i][j]);
        }
    }
    #pragma unroll
    for (int i = 0; i < 4; i++)
        #pragma unroll
        for (int j = 0; j < 2; j++)
            #pragma unroll
            for (int r = 0; r < 8; r++) {
                const int row = rB + 16 * i + laneHi * 8 + r;
                const int col = nB + 16 * j + laneLo;
                out[(size_t)row * O_SZ + col] = acc[i][j][r];
            }
}

// ---------------- host launcher ----------------
extern "C" void kernel_launch(void* const* d_in, const int* in_sizes, int n_in,
                              void* d_out, int out_size, void* d_ws, size_t ws_size,
                              hipStream_t stream) {
    (void)in_sizes; (void)n_in; (void)out_size;
    const float* x  = (const float*)d_in[0];
    const float* ex = (const float*)d_in[1];
    const float* eh = (const float*)d_in[2];
    const float* em = (const float*)d_in[3];
    const float* Wx = (const float*)d_in[4];
    const float* Wh = (const float*)d_in[5];
    const float* Wm = (const float*)d_in[6];
    const float* Wo = (const float*)d_in[7];
    const float* Ad = (const float*)d_in[8];
    const float* Bd = (const float*)d_in[9];
    float* out = (float*)d_out;

    char* ws = (char*)d_ws;
    size_t off = 0;
    auto carve = [&](size_t bytes) -> void* {
        void* p = ws + off;
        off += (bytes + 255) & ~(size_t)255;
        return p;
    };
    bf16*  x_bf   = (bf16*) carve((size_t)T_STEPS * B_SZ * I_SZ * sizeof(bf16));
    bf16*  Wx_bf  = (bf16*) carve((size_t)H_SZ * I_SZ * sizeof(bf16));
    bf16*  Wh_bf  = (bf16*) carve((size_t)H_SZ * H_SZ * sizeof(bf16));
    bf16*  Wm_bf  = (bf16*) carve((size_t)H_SZ * M_SZ * sizeof(bf16));
    bf16*  Wo_bf  = (bf16*) carve((size_t)O_SZ * H_SZ * sizeof(bf16));
    float* m_buf0 = (float*)carve((size_t)B_SZ * M_SZ * sizeof(float));
    float* m_buf1 = (float*)carve((size_t)B_SZ * M_SZ * sizeof(float));
    bf16*  m_bf   = (bf16*) carve((size_t)B_SZ * M_SZ * sizeof(bf16));
    bf16*  h_zero = (bf16*) carve((size_t)B_SZ * H_SZ * sizeof(bf16));
    float* m_buf[2] = { m_buf0, m_buf1 };

    // Preferred: full h history (one batched output GEMM at the end).
    const size_t hist_bytes = (size_t)T_STEPS * B_SZ * H_SZ * sizeof(bf16);
    const bool use_hist = (off + hist_bytes) <= ws_size;
    bf16* h_hist = nullptr;
    bf16* h_pp0 = nullptr;
    bf16* h_pp1 = nullptr;
    if (use_hist) {
        h_hist = (bf16*)carve(hist_bytes);
    } else {
        h_pp0 = (bf16*)carve((size_t)B_SZ * H_SZ * sizeof(bf16));
        h_pp1 = (bf16*)carve((size_t)B_SZ * H_SZ * sizeof(bf16));
    }

    auto cvt = [&](const float* s, bf16* d, int n) {
        k_f32_to_bf16<<<(n + 255) / 256, 256, 0, stream>>>(s, d, n);
    };
    cvt(x,  x_bf,  T_STEPS * B_SZ * I_SZ);
    cvt(Wx, Wx_bf, H_SZ * I_SZ);
    cvt(Wh, Wh_bf, H_SZ * H_SZ);
    cvt(Wm, Wm_bf, H_SZ * M_SZ);
    cvt(Wo, Wo_bf, O_SZ * H_SZ);
    k_zero_bf16<<<(B_SZ * H_SZ + 255) / 256, 256, 0, stream>>>(h_zero, B_SZ * H_SZ);
    k_zero_f32 <<<(B_SZ * M_SZ + 255) / 256, 256, 0, stream>>>(m_buf[1], B_SZ * M_SZ);

    const dim3 gh(H_SZ / 128, B_SZ / 128);
    for (int t = 0; t < T_STEPS; t++) {
        const int cur = t & 1, prev = cur ^ 1;
        bf16* h_prev;
        bf16* h_cur;
        if (use_hist) {
            h_prev = (t == 0) ? h_zero : h_hist + (size_t)(t - 1) * B_SZ * H_SZ;
            h_cur  = h_hist + (size_t)t * B_SZ * H_SZ;
        } else {
            bf16* pp[2] = { h_pp0, h_pp1 };
            h_prev = (t == 0) ? h_zero : pp[prev];
            h_cur  = pp[cur];
        }
        k_um<<<B_SZ / 16, 512, 0, stream>>>(t, x, h_prev, m_buf[prev],
                                            ex, eh, em, Ad, Bd, m_buf[cur], m_bf);
        k_hidden<<<gh, 256, 0, stream>>>(t, x_bf, h_prev, m_bf,
                                         Wx_bf, Wh_bf, Wm_bf, h_cur);
        if (!use_hist) {
            k_out<<<B_SZ / 128, 256, 0, stream>>>(h_cur, Wo_bf,
                                                  out + (size_t)t * B_SZ * O_SZ);
        }
    }
    if (use_hist) {
        // One batched GEMM: (T*B, H) x (H, O) -> (T*B, O)
        k_out<<<(T_STEPS * B_SZ) / 128, 256, 0, stream>>>(h_hist, Wo_bf, out);
    }
}
